// CustomLSTMCell_2052994367705
// MI455X (gfx1250) — compile-verified, matrix operand order checked by reading, not run on hardware
//
#include <hip/hip_runtime.h>

// ---------------------------------------------------------------------------
// LSTM cell fused kernel for gfx1250 (MI455X), wave32 + WMMA bf16.
//   combined = [x | h_prev]  (8192 x 1024)
//   gate_g   = combined @ W_g^T + b_g   (8192 x 512), g in {i,f,c,o}
//   c_next = sigm(f)*c_prev + sigm(i)*tanh(c~);  h_next = sigm(o)*tanh(c_next)
//
// Mapping note: all 8 waves of a block share one 16-col nTile, so the four
// W-fragment loads per K-step are the SAME addresses wave-to-wave -> WGP$
// broadcast hits (W traffic ~4x lower than the nTile-per-wave mapping).
// Consecutive blocks vary nTile first, so A rows stay hot in the 192MB L2.
// ---------------------------------------------------------------------------

typedef __attribute__((ext_vector_type(16))) __bf16 v16bf;
typedef __attribute__((ext_vector_type(8)))  __bf16 v8bf;
typedef __attribute__((ext_vector_type(8)))  float  v8f;

#define BATCH_N 8192
#define KDIM    1024   // INPUT_DIM + HIDDEN_DIM
#define HID     512

// ---- f32 -> bf16 round-to-nearest-even --------------------------------------
__device__ __forceinline__ unsigned short f2bf(float f) {
    union { float f; unsigned u; } v; v.f = f;
    unsigned u = v.u;
    u += 0x7FFFu + ((u >> 16) & 1u);
    return (unsigned short)(u >> 16);
}

// Convert & concatenate [x | h_prev] -> bf16 A (BATCH_N x KDIM), K-contiguous.
__global__ __launch_bounds__(256)
void lstm_convert_A(const float* __restrict__ x, const float* __restrict__ h,
                    unsigned short* __restrict__ A) {
    size_t tid  = (size_t)blockIdx.x * blockDim.x + threadIdx.x;
    size_t idx4 = tid * 4;
    if (idx4 >= (size_t)BATCH_N * KDIM) return;
    size_t b = idx4 >> 10;             // row
    int    j = (int)(idx4 & 1023);     // col in combined
    const float* src = (j < 512) ? (x + b * 512 + j)
                                 : (h + b * 512 + (j - 512));
    float4 v = *(const float4*)src;
    ushort4 o;
    o.x = f2bf(v.x); o.y = f2bf(v.y); o.z = f2bf(v.z); o.w = f2bf(v.w);
    *(ushort4*)(A + idx4) = o;
}

// Convert 4 weight matrices -> packed bf16 (4 x HID x KDIM), K-contiguous.
__global__ __launch_bounds__(256)
void lstm_convert_W(const float* __restrict__ Wi, const float* __restrict__ Wf,
                    const float* __restrict__ Wc, const float* __restrict__ Wo,
                    unsigned short* __restrict__ Wp) {
    size_t tid  = (size_t)blockIdx.x * blockDim.x + threadIdx.x;
    size_t idx4 = tid * 4;
    const size_t per_gate = (size_t)HID * KDIM;          // 2^19
    if (idx4 >= 4 * per_gate) return;
    int    g   = (int)(idx4 >> 19);
    size_t rem = idx4 & (per_gate - 1);
    const float* srcs[4] = { Wi, Wf, Wc, Wo };
    float4 v = *(const float4*)(srcs[g] + rem);
    ushort4 o;
    o.x = f2bf(v.x); o.y = f2bf(v.y); o.z = f2bf(v.z); o.w = f2bf(v.w);
    *(ushort4*)(Wp + idx4) = o;
}

// Load one 16-element bf16 fragment: K {base..base+7} and {base+16..base+23}
// as two aligned b128 loads (matches 16-bit A-matrix VGPR layout, ISA 7.12.2).
__device__ __forceinline__ v16bf ld_frag(const unsigned short* p) {
    v8bf lo = *(const v8bf*)(p);
    v8bf hi = *(const v8bf*)(p + 16);
    v16bf r;
#pragma unroll
    for (int i = 0; i < 8; ++i) { r[i] = lo[i]; r[i + 8] = hi[i]; }
    return r;
}

__device__ __forceinline__ float sigm_f(float x) {
    return 1.0f / (1.0f + __expf(-x));
}
__device__ __forceinline__ float tanh_f(float x) {
    return 2.0f / (1.0f + __expf(-2.0f * x)) - 1.0f;
}

// Fused 4-gate GEMM + LSTM epilogue.
// Block tile: 256 batch rows x 16 hidden cols. 8 waves, each 32x16 x 4 gates.
__global__ __launch_bounds__(256)
void lstm_wmma_kernel(const unsigned short* __restrict__ A,
                      const unsigned short* __restrict__ W,
                      const float* __restrict__ pb_i, const float* __restrict__ pb_f,
                      const float* __restrict__ pb_c, const float* __restrict__ pb_o,
                      const float* __restrict__ c_prev,
                      float* __restrict__ h_out, float* __restrict__ c_out) {
    const int lane   = threadIdx.x & 31;
    const int laneLo = lane & 15;
    const int laneHi = lane >> 4;
    const int kbase  = laneHi * 8;      // K phase per half-wave (ISA layout)
    const int wave   = threadIdx.x >> 5;

    // All waves in this block share nTile (identical W fragment addresses);
    // each wave takes a distinct 32-row slice of the 256-row block tile.
    const int nTile  = blockIdx.x & 31;        // varies fastest across blocks
    const int mBlock = blockIdx.x >> 5;        // 32 blocks of 256 rows
    const int mBase  = mBlock * 256 + wave * 32;
    const int nBase  = nTile * 16;

    const unsigned short* a0 = A + (size_t)(mBase + laneLo) * KDIM + kbase;
    const unsigned short* a1 = a0 + (size_t)16 * KDIM;  // rows mBase+16..+31
    const unsigned short* wp[4];
#pragma unroll
    for (int g = 0; g < 4; ++g)
        wp[g] = W + ((size_t)g * HID + nBase + laneLo) * KDIM + kbase;

    v8f acc[2][4];
#pragma unroll
    for (int s = 0; s < 2; ++s)
#pragma unroll
        for (int g = 0; g < 4; ++g)
#pragma unroll
            for (int r = 0; r < 8; ++r) acc[s][g][r] = 0.0f;

#pragma unroll 4
    for (int k0 = 0; k0 < KDIM; k0 += 32) {
        // Prefetch A stream ahead (global_prefetch_b8).
        __builtin_prefetch((const void*)(a0 + k0 + 256), 0, 1);
        __builtin_prefetch((const void*)(a1 + k0 + 256), 0, 1);

        v16bf va0 = ld_frag(a0 + k0);
        v16bf va1 = ld_frag(a1 + k0);
#pragma unroll
        for (int g = 0; g < 4; ++g) {
            v16bf vb = ld_frag(wp[g] + k0);
            acc[0][g] = __builtin_amdgcn_wmma_f32_16x16x32_bf16(
                false, va0, false, vb, (short)0, acc[0][g], false, false);
            acc[1][g] = __builtin_amdgcn_wmma_f32_16x16x32_bf16(
                false, va1, false, vb, (short)0, acc[1][g], false, false);
        }
    }

    // ---- Epilogue: bias + gates + cell update -------------------------------
    const int col = nBase + laneLo;                 // C/D layout: N = lane%16
    const float bi = pb_i[col];
    const float bf = pb_f[col];
    const float bc = pb_c[col];
    const float bo = pb_o[col];

#pragma unroll
    for (int s = 0; s < 2; ++s) {
        const int rowBase = mBase + s * 16 + laneHi * 8;  // M = vgpr + 8*(lane/16)
#pragma unroll
        for (int r = 0; r < 8; ++r) {
            const size_t idx = (size_t)(rowBase + r) * HID + col;
            const float gi = sigm_f(acc[s][0][r] + bi);
            const float gf = sigm_f(acc[s][1][r] + bf);
            const float gc = tanh_f(acc[s][2][r] + bc);
            const float go = sigm_f(acc[s][3][r] + bo);
            const float cp = c_prev[idx];
            const float cn = gf * cp + gi * gc;
            h_out[idx] = go * tanh_f(cn);
            c_out[idx] = cn;
        }
    }
}

extern "C" void kernel_launch(void* const* d_in, const int* in_sizes, int n_in,
                              void* d_out, int out_size, void* d_ws, size_t ws_size,
                              hipStream_t stream) {
    // setup_inputs order: x, h_prev, c_prev, W_i, b_i, W_f, b_f, W_c, b_c, W_o, b_o
    const float* x      = (const float*)d_in[0];
    const float* h_prev = (const float*)d_in[1];
    const float* c_prev = (const float*)d_in[2];
    const float* W_i    = (const float*)d_in[3];
    const float* b_i    = (const float*)d_in[4];
    const float* W_f    = (const float*)d_in[5];
    const float* b_f    = (const float*)d_in[6];
    const float* W_c    = (const float*)d_in[7];
    const float* b_c    = (const float*)d_in[8];
    const float* W_o    = (const float*)d_in[9];
    const float* b_o    = (const float*)d_in[10];

    // Workspace layout: bf16 A (16 MB) then bf16 packed W (4 MB).
    unsigned short* Abf = (unsigned short*)d_ws;
    unsigned short* Wbf = Abf + (size_t)BATCH_N * KDIM;

    {   // A: 8192*1024 / 4 elems per thread = 2M threads
        const int threads = 256;
        const int blocks  = (BATCH_N * KDIM / 4 + threads - 1) / threads; // 8192
        lstm_convert_A<<<blocks, threads, 0, stream>>>(x, h_prev, Abf);
    }
    {   // W: 4*512*1024 / 4 = 512K threads
        const int threads = 256;
        const int blocks  = (4 * HID * KDIM / 4 + threads - 1) / threads; // 2048
        lstm_convert_W<<<blocks, threads, 0, stream>>>(W_i, W_f, W_c, W_o, Wbf);
    }

    float* h_out = (float*)d_out;                 // outputs: h_next then c_next
    float* c_out = h_out + (size_t)BATCH_N * HID;

    // (8192/256 row-blocks) * (512/16 col-tiles) = 1024 blocks, 8 waves each.
    const int gemm_blocks = (BATCH_N / 256) * (HID / 16);  // 1024
    lstm_wmma_kernel<<<gemm_blocks, 256, 0, stream>>>(
        Abf, Wbf, b_i, b_f, b_c, b_o, c_prev, h_out, c_out);
}
